// JointEncoder_47081431499479
// MI455X (gfx1250) — compile-verified, matrix operand order checked by reading, not run on hardware
//
#include <hip/hip_runtime.h>
#include <stdint.h>

typedef _Float16 f16;
typedef __attribute__((ext_vector_type(16))) _Float16 v16h;
typedef __attribute__((ext_vector_type(8)))  float    v8f;

#define B_    4
#define N_    4096
#define K1_   4093
#define K2_   1351
#define MAXNB_ 64

// ---------------- weight convert: f32 row-major -> f16 WMMA-fragment-swizzled ----------------
// Output layout: tile t = kt*NT + nt, then [lane][i] with 16 contiguous halfs per lane.
// Lane semantics match the CDNA5 16-bit B-operand layout used by the loader below:
//   col = nt*16 + (lane&15), kh = (lane>>4)*8,
//   i<8  -> k = kt*32 + kh + i
//   i>=8 -> k = kt*32 + kh + 16 + (i-8)
__global__ void convert_swizzle_kernel(const float* __restrict__ W, f16* __restrict__ out,
                                       int cin, int cout, int cinp) {
  int total = cinp * cout;           // == (cinp/32)*(cout/16)*512
  int NT = cout >> 4;
  for (int t = blockIdx.x * blockDim.x + threadIdx.x; t < total; t += gridDim.x * blockDim.x) {
    int i    = t & 15;
    int lane = (t >> 4) & 31;
    int tile = t >> 9;
    int nt = tile % NT;
    int kt = tile / NT;
    int col = nt * 16 + (lane & 15);
    int kh  = (lane >> 4) * 8;
    int k   = kt * 32 + kh + ((i < 8) ? i : (8 + i));
    out[t] = (k < cin) ? (f16)W[k * cout + col] : (f16)0.0f;
  }
}

// ---------------- farthest point sampling (one workgroup per cloud) ----------------
__global__ void fps_kernel(const float* __restrict__ pos, int n, int k, int* __restrict__ idx_out) {
  const int b = blockIdx.x;
  const float* p = pos + (size_t)b * n * 3;
  extern __shared__ char sm[];
  float* px = (float*)sm;
  float* py = px + n;
  float* pz = py + n;
  float* dist = pz + n;
  float* rd = dist + n;
  int*   ri = (int*)(rd + blockDim.x);
  const int tid = threadIdx.x, bs = blockDim.x;

  for (int j = tid; j < n; j += bs) { px[j] = p[j*3+0]; py[j] = p[j*3+1]; pz[j] = p[j*3+2]; }
  __syncthreads();
  const float x0 = px[0], y0 = py[0], z0 = pz[0];
  for (int j = tid; j < n; j += bs) {
    float dx = px[j]-x0, dy = py[j]-y0, dz = pz[j]-z0;
    dist[j] = dx*dx + dy*dy + dz*dz;
  }
  if (tid == 0) idx_out[(size_t)b * k] = 0;
  __syncthreads();
  for (int it = 1; it < k; ++it) {
    float bd = -1.0f; int bi = n;
    for (int j = tid; j < n; j += bs) { float d = dist[j]; if (d > bd) { bd = d; bi = j; } }
    rd[tid] = bd; ri[tid] = bi;
    __syncthreads();
    for (int s = bs >> 1; s > 0; s >>= 1) {
      if (tid < s) {
        float od = rd[tid+s]; int oi = ri[tid+s];
        if (od > rd[tid] || (od == rd[tid] && oi < ri[tid])) { rd[tid] = od; ri[tid] = oi; }
      }
      __syncthreads();
    }
    int best = ri[0];
    if (tid == 0) idx_out[(size_t)b * k + it] = best;
    float bx = px[best], by = py[best], bz = pz[best];
    for (int j = tid; j < n; j += bs) {
      float dx = px[j]-bx, dy = py[j]-by, dz = pz[j]-bz;
      dist[j] = fminf(dist[j], dx*dx + dy*dy + dz*dz);
    }
    __syncthreads();
  }
}

// ---------------- gather sampled positions ----------------
__global__ void gather_pos_kernel(const float* __restrict__ pos, const int* __restrict__ idx,
                                  float* __restrict__ qpos, int n, int k, int total) {
  int t = blockIdx.x * blockDim.x + threadIdx.x;
  if (t >= total) return;
  int b = t / k;
  int j = idx[t];
  const float* s = pos + ((size_t)b * n + j) * 3;
  float* d = qpos + (size_t)t * 3;
  d[0] = s[0]; d[1] = s[1]; d[2] = s[2];
}

// ---------------- ball query: one wave32 per query, index-ordered first-64 ----------------
__global__ void ball_query_kernel(const float* __restrict__ src, const float* __restrict__ q,
                                  int n, int k, float r2,
                                  int* __restrict__ nb, int* __restrict__ cnt, int total) {
  int w = (int)((blockIdx.x * blockDim.x + threadIdx.x) >> 5);
  int lane = threadIdx.x & 31;
  if (w >= total) return;
  int b = w / k;
  const float* sp = src + (size_t)b * n * 3;
  float qx = q[(size_t)w*3+0], qy = q[(size_t)w*3+1], qz = q[(size_t)w*3+2];
  int* nbp = nb + (size_t)w * MAXNB_;
  int c = 0;
  for (int base = 0; base < n && c < MAXNB_; base += 32) {
    int j = base + lane;
    bool valid = false;
    if (j < n) {
      float dx = sp[j*3+0]-qx, dy = sp[j*3+1]-qy, dz = sp[j*3+2]-qz;
      valid = (dx*dx + dy*dy + dz*dz) <= r2;
    }
    unsigned m = (unsigned)__ballot(valid);
    int before = __popc(m & ((1u << lane) - 1u));
    int slot = c + before;
    if (valid && slot < MAXNB_) nbp[slot] = j;
    c += __popc(m);
  }
  if (c > MAXNB_) c = MAXNB_;
  for (int s = c + lane; s < MAXNB_; s += 32) nbp[s] = 0;
  if (lane == 0) cnt[w] = c;
}

// ---------------- one MLP layer: [64 x LDA] (LDS,f16) @ swizzled weights (global,f16) ----------------
// Each wave computes one 16-row M-tile; N blocked by 4 tiles so each A fragment feeds 4 WMMAs.
// B fragments are single 32-byte contiguous v16h loads (pre-swizzled by convert_swizzle_kernel).
template<int LDA, int KTILES, int COUT>
__device__ __forceinline__ void wmma_layer(const f16* A, const f16* __restrict__ W,
                                           const float* __restrict__ bias, f16* H,
                                           int wid, int lane) {
  constexpr int NT = COUT / 16;
  static_assert(NT % 4 == 0, "COUT must be a multiple of 64");
  const int kh   = (lane >> 4) * 8;
  const int coln = lane & 15;
  const f16* Arow = A + (size_t)(wid * 16 + (lane & 15)) * LDA + kh;
  const int row0 = wid * 16 + (lane >> 4) * 8;
  __builtin_prefetch(W, 0, 3);
  for (int ntb = 0; ntb < NT; ntb += 4) {
    v8f acc[4];
#pragma unroll
    for (int u = 0; u < 4; ++u) acc[u] = (v8f){0.f,0.f,0.f,0.f,0.f,0.f,0.f,0.f};
#pragma unroll
    for (int kt = 0; kt < KTILES; ++kt) {
      v16h a;
      const f16* ab = Arow + kt * 32;
#pragma unroll
      for (int i = 0; i < 8; ++i) { a[i] = ab[i]; a[8 + i] = ab[16 + i]; }
      const f16* wt = W + ((size_t)(kt * NT + ntb) * 32 + lane) * 16;
#pragma unroll
      for (int u = 0; u < 4; ++u) {
        v16h bf = *(const v16h*)(wt + (size_t)u * 32 * 16);
        acc[u] = __builtin_amdgcn_wmma_f32_16x16x32_f16(false, a, false, bf, (short)0, acc[u], false, false);
      }
    }
#pragma unroll
    for (int u = 0; u < 4; ++u) {
      float bv = bias[(ntb + u) * 16 + coln];
#pragma unroll
      for (int r = 0; r < 8; ++r) {
        float v = acc[u][r] + bv;
        H[(size_t)(row0 + r) * COUT + (ntb + u) * 16 + coln] = (f16)(v > 0.f ? v : 0.f);
      }
    }
  }
}

// ---------------- SA1: rel(3) -> 64 -> 64 -> 128, masked max over 64 neighbors ----------------
__global__ void __launch_bounds__(128) sa1_kernel(const float* __restrict__ pos, const float* __restrict__ qpos,
    const int* __restrict__ nb, const int* __restrict__ cnt,
    const f16* __restrict__ W0, const f16* __restrict__ W1, const f16* __restrict__ W2,
    const float* __restrict__ b0, const float* __restrict__ b1, const float* __restrict__ b2,
    f16* __restrict__ x1) {
  const int q = blockIdx.x;
  const int b = q / K1_;
  extern __shared__ char sm[];
  f16* A0 = (f16*)sm;          // 64*32
  f16* H1 = A0 + 64 * 32;      // 64*64
  f16* H2 = H1 + 64 * 64;      // 64*64
  f16* H3 = H2 + 64 * 64;      // 64*128
  const int tid = threadIdx.x, wid = tid >> 5, lane = tid & 31;
  const int* nbp = nb + (size_t)q * MAXNB_;
  const float qx = qpos[(size_t)q*3+0], qy = qpos[(size_t)q*3+1], qz = qpos[(size_t)q*3+2];
  for (int e = tid; e < 64 * 32; e += 128) {
    int row = e >> 5, col = e & 31;
    f16 v = (f16)0.0f;
    if (col < 3) {
      int j = nbp[row];
      float pv = pos[((size_t)b * N_ + j) * 3 + col];
      float qv = (col == 0) ? qx : (col == 1) ? qy : qz;
      v = (f16)(pv - qv);
    }
    A0[e] = v;
  }
  __syncthreads();
  wmma_layer<32, 1, 64>(A0, W0, b0, H1, wid, lane);  __syncthreads();
  wmma_layer<64, 2, 64>(H1, W1, b1, H2, wid, lane);  __syncthreads();
  wmma_layer<64, 2, 128>(H2, W2, b2, H3, wid, lane); __syncthreads();
  const int c = cnt[q];
  for (int ch = tid; ch < 128; ch += 128) {
    float m = -3.4e38f;
    for (int r = 0; r < c; ++r) m = fmaxf(m, (float)H3[r * 128 + ch]);
    x1[(size_t)q * 128 + ch] = (f16)m;
  }
}

// ---------------- SA2: [gx(128), rel(3)] -> 128 -> 128 -> 256 ----------------
__global__ void __launch_bounds__(128) sa2_kernel(const f16* __restrict__ x1, const float* __restrict__ p1,
    const float* __restrict__ p2, const int* __restrict__ nb, const int* __restrict__ cnt,
    const f16* __restrict__ W0, const f16* __restrict__ W1, const f16* __restrict__ W2,
    const float* __restrict__ b0, const float* __restrict__ b1, const float* __restrict__ b2,
    f16* __restrict__ x2) {
  const int q = blockIdx.x;
  const int b = q / K2_;
  extern __shared__ char sm[];
  f16* A0 = (f16*)sm;          // 64*160 = 10240 halfs
  f16* H1 = A0 + 64 * 160;     // 64*128 = 8192
  f16* H2 = H1 + 64 * 128;     // 64*128 = 8192
  f16* H3 = A0;                // alias: 64*256 = 16384 halfs, overlays dead A0+H1
  const int tid = threadIdx.x, wid = tid >> 5, lane = tid & 31;
  const int* nbp = nb + (size_t)q * MAXNB_;
  for (int e = tid; e < 64 * 160; e += 128) {
    int row = e / 160, col = e - row * 160;
    int j = nbp[row];
    f16 v = (f16)0.0f;
    if (col < 128) {
      v = x1[((size_t)b * K1_ + j) * 128 + col];
    } else if (col < 131) {
      int cc = col - 128;
      v = (f16)(p1[((size_t)b * K1_ + j) * 3 + cc] - p2[(size_t)q * 3 + cc]);
    }
    A0[e] = v;
  }
  __syncthreads();
  wmma_layer<160, 5, 128>(A0, W0, b0, H1, wid, lane); __syncthreads();
  wmma_layer<128, 4, 128>(H1, W1, b1, H2, wid, lane); __syncthreads();
  wmma_layer<128, 4, 256>(H2, W2, b2, H3, wid, lane); __syncthreads();
  const int c = cnt[q];
  for (int ch = tid; ch < 256; ch += 128) {
    float m = -3.4e38f;
    for (int r = 0; r < c; ++r) m = fmaxf(m, (float)H3[r * 256 + ch]);
    x2[(size_t)q * 256 + ch] = (f16)m;
  }
}

// ---------------- SA3: per-point [x2(256), p2(3)] -> 256 -> 256 -> 512 -> 256 -> 128 ----------------
__global__ void __launch_bounds__(128) sa3_kernel(const f16* __restrict__ x2, const float* __restrict__ p2,
    const f16* W0, const f16* W1, const f16* W2, const f16* W3, const f16* W4,
    const float* b0, const float* b1, const float* b2, const float* b3, const float* b4,
    float* __restrict__ h5, int total) {
  const int p0 = blockIdx.x * 64;
  extern __shared__ char sm[];
  f16* buf0 = (f16*)sm;          // 64*512
  f16* buf1 = buf0 + 64 * 512;   // 64*512
  const int tid = threadIdx.x, wid = tid >> 5, lane = tid & 31;
  for (int e = tid; e < 64 * 288; e += 128) {
    int row = e / 288, col = e - row * 288;
    int p = p0 + row;
    f16 v = (f16)0.0f;
    if (p < total) {
      if (col < 256)      v = x2[(size_t)p * 256 + col];
      else if (col < 259) v = (f16)p2[(size_t)p * 3 + (col - 256)];
    }
    buf0[e] = v;
  }
  __syncthreads();
  wmma_layer<288, 9, 256>(buf0, W0, b0, buf1, wid, lane); __syncthreads();
  wmma_layer<256, 8, 256>(buf1, W1, b1, buf0, wid, lane); __syncthreads();
  wmma_layer<256, 8, 512>(buf0, W2, b2, buf1, wid, lane); __syncthreads();
  wmma_layer<512, 16, 256>(buf1, W3, b3, buf0, wid, lane); __syncthreads();
  wmma_layer<256, 8, 128>(buf0, W4, b4, buf1, wid, lane); __syncthreads();
  for (int e = tid; e < 64 * 128; e += 128) {
    int row = e >> 7, ch = e & 127;
    int p = p0 + row;
    if (p < total) h5[(size_t)p * 128 + ch] = (float)buf1[row * 128 + ch];
  }
}

// ---------------- global max over K2 points ----------------
__global__ void final_max_kernel(const float* __restrict__ h5, float* __restrict__ out) {
  const int b = blockIdx.x, ch = threadIdx.x;
  float m = -3.4e38f;
  for (int p = 0; p < K2_; ++p) m = fmaxf(m, h5[((size_t)b * K2_ + p) * 128 + ch]);
  out[b * 128 + ch] = m;
}

extern "C" void kernel_launch(void* const* d_in, const int* in_sizes, int n_in,
                              void* d_out, int out_size, void* d_ws, size_t ws_size,
                              hipStream_t stream) {
  (void)out_size; (void)ws_size;
  const float* joints = (const float*)d_in[0];
  int wb = 1;
  if (n_in >= 3 && in_sizes[1] == B_ * N_) wb = 2;  // joints_batch flattened second
  const float* Wf[11]; const float* Bf[11];
  for (int i = 0; i < 11; ++i) { Wf[i] = (const float*)d_in[wb + 2*i]; Bf[i] = (const float*)d_in[wb + 2*i + 1]; }

  char* ws = (char*)d_ws;
  size_t off = 0;
  auto alloc = [&](size_t bytes) -> void* {
    void* p = ws + off; off += (bytes + 255) & ~(size_t)255; return p;
  };

  // f16 weights, K padded to multiple of 32, swizzled to WMMA fragment order
  struct Dim { int cin, cout, cinp; };
  const Dim cv[11] = {
    {3, 64, 32}, {64, 64, 64}, {64, 128, 64},
    {131, 128, 160}, {128, 128, 128}, {128, 256, 128},
    {259, 256, 288}, {256, 256, 256}, {256, 512, 256}, {512, 256, 512}, {256, 128, 256}
  };
  f16* Wh[11];
  for (int i = 0; i < 11; ++i) {
    int total = cv[i].cinp * cv[i].cout;
    Wh[i] = (f16*)alloc((size_t)total * sizeof(f16));
    int grid = (total + 255) / 256;
    convert_swizzle_kernel<<<grid, 256, 0, stream>>>(Wf[i], Wh[i], cv[i].cin, cv[i].cout, cv[i].cinp);
  }

  int*   idx1  = (int*)alloc((size_t)B_ * K1_ * 4);
  float* qpos1 = (float*)alloc((size_t)B_ * K1_ * 3 * 4);
  int*   nb1   = (int*)alloc((size_t)B_ * K1_ * MAXNB_ * 4);
  int*   cnt1  = (int*)alloc((size_t)B_ * K1_ * 4);
  f16*   x1    = (f16*)alloc((size_t)B_ * K1_ * 128 * 2);
  int*   idx2  = (int*)alloc((size_t)B_ * K2_ * 4);
  float* qpos2 = (float*)alloc((size_t)B_ * K2_ * 3 * 4);
  int*   nb2   = (int*)alloc((size_t)B_ * K2_ * MAXNB_ * 4);
  int*   cnt2  = (int*)alloc((size_t)B_ * K2_ * 4);
  f16*   x2    = (f16*)alloc((size_t)B_ * K2_ * 256 * 2);
  float* h5    = (float*)alloc((size_t)B_ * K2_ * 128 * 4);

  // ---- Level 1 ----
  fps_kernel<<<B_, 256, 16 * N_ + 256 * 8, stream>>>(joints, N_, K1_, idx1);
  gather_pos_kernel<<<(B_ * K1_ + 255) / 256, 256, 0, stream>>>(joints, idx1, qpos1, N_, K1_, B_ * K1_);
  ball_query_kernel<<<(B_ * K1_ + 7) / 8, 256, 0, stream>>>(joints, qpos1, N_, K1_, 0.16f, nb1, cnt1, B_ * K1_);
  sa1_kernel<<<B_ * K1_, 128, (64*32 + 64*64 + 64*64 + 64*128) * 2, stream>>>(
      joints, qpos1, nb1, cnt1, Wh[0], Wh[1], Wh[2], Bf[0], Bf[1], Bf[2], x1);

  // ---- Level 2 ----
  fps_kernel<<<B_, 256, 16 * K1_ + 256 * 8, stream>>>(qpos1, K1_, K2_, idx2);
  gather_pos_kernel<<<(B_ * K2_ + 255) / 256, 256, 0, stream>>>(qpos1, idx2, qpos2, K1_, K2_, B_ * K2_);
  ball_query_kernel<<<(B_ * K2_ + 7) / 8, 256, 0, stream>>>(qpos1, qpos2, K1_, K2_, 0.36f, nb2, cnt2, B_ * K2_);
  sa2_kernel<<<B_ * K2_, 128, (64*160 + 64*128 + 64*128) * 2, stream>>>(
      x1, qpos1, qpos2, nb2, cnt2, Wh[3], Wh[4], Wh[5], Bf[3], Bf[4], Bf[5], x2);

  // ---- Level 3 + global pool ----
  sa3_kernel<<<(B_ * K2_ + 63) / 64, 128, 2 * 64 * 512 * 2, stream>>>(
      x2, qpos2, Wh[6], Wh[7], Wh[8], Wh[9], Wh[10],
      Bf[6], Bf[7], Bf[8], Bf[9], Bf[10], h5, B_ * K2_);
  final_max_kernel<<<B_, 128, 0, stream>>>(h5, (float*)d_out);
}
